// MoE_32246614459304
// MI455X (gfx1250) — matrix operand order repaired and checked by reading, not compiled, not run on hardware
//
#include <hip/hip_runtime.h>

#define B_TOK 16384
#define D_DIM 128
#define H_DIM 512
#define E_NUM 8
#define BD (B_TOK * D_DIM)
#define M_TILE 32                    // tokens per k_moe block (2 WMMA M-tiles)
#define TPE (B_TOK / M_TILE)         // tiles per expert (capacity = all tokens)
#define EPS_COMBINE 2.220446049250313e-16f

typedef __attribute__((ext_vector_type(16))) __bf16 v16bf;
typedef __attribute__((ext_vector_type(8)))  float  v8f;

union FragU { unsigned u[8]; v16bf v; };

__device__ __forceinline__ unsigned short f2bf(float f) {
    union { float f; unsigned u; } x; x.f = f;
    unsigned r = x.u + 0x7FFFu + ((x.u >> 16) & 1u);   // round-to-nearest-even
    return (unsigned short)(r >> 16);
}
__device__ __forceinline__ unsigned pack2(float lo, float hi) {
    return (unsigned)f2bf(lo) | ((unsigned)f2bf(hi) << 16);
}

// ---------------- init: zero y, counters, importance ----------------
__global__ void k_zero(float* y, int* cnt, float* imp) {
    int i = blockIdx.x * blockDim.x + threadIdx.x;
    if (i < BD) y[i] = 0.0f;
    if (i < E_NUM) { cnt[i] = 0; imp[i] = 0.0f; }
}

// ---------------- pack x (fp32 -> bf16 pairs, row-major) ----------------
__global__ void k_pack_x(const float* __restrict__ x, unsigned* __restrict__ xbf) {
    int i = blockIdx.x * blockDim.x + threadIdx.x;      // < BD/2
    xbf[i] = pack2(x[2 * i], x[2 * i + 1]);
}

// ---- pack W1 into fragment-major layout: [e][nt(32)][ks(4)][lane(32)][i(8)] ----
// B-fragment (16-bit, 32x16): lane holds col N = lane%16,
// K = 32*ks + (lane>=16 ? 16 : 0) + 2*i (+1 in high half of dword)
__global__ void k_pack_w1(const float* __restrict__ W1, unsigned* __restrict__ W1p) {
    int idx = blockIdx.x * blockDim.x + threadIdx.x;    // < 8*32*4*32*8 = 262144
    int i8   =  idx        & 7;
    int lane = (idx >> 3)  & 31;
    int ks   = (idx >> 8)  & 3;
    int nt   = (idx >> 10) & 31;
    int e    =  idx >> 15;
    int K = ks * 32 + ((lane >> 4) & 1) * 16 + 2 * i8;
    int N = nt * 16 + (lane & 15);
    const float* p = W1 + ((size_t)(e * D_DIM + K) * H_DIM + N);
    W1p[idx] = pack2(p[0], p[H_DIM]);
}

// ---- pack W2: [e][nt(8)][ks(16)][lane(32)][i(8)] ----
__global__ void k_pack_w2(const float* __restrict__ W2, unsigned* __restrict__ W2p) {
    int idx = blockIdx.x * blockDim.x + threadIdx.x;    // < 8*8*16*32*8 = 262144
    int i8   =  idx        & 7;
    int lane = (idx >> 3)  & 31;
    int ks   = (idx >> 8)  & 15;
    int nt   = (idx >> 12) & 7;
    int e    =  idx >> 15;
    int K = ks * 32 + ((lane >> 4) & 1) * 16 + 2 * i8;
    int N = nt * 16 + (lane & 15);
    const float* p = W2 + ((size_t)(e * H_DIM + K) * D_DIM + N);
    W2p[idx] = pack2(p[0], p[D_DIM]);
}

// ---------------- gating: logits, top-2, softmax, expert lists ----------------
__global__ void k_gate(const float* __restrict__ x, const float* __restrict__ wg,
                       int* __restrict__ cnt, float* __restrict__ imp,
                       int* __restrict__ idxl, float* __restrict__ gatel) {
    __shared__ float swg[D_DIM * E_NUM];
    int tid = threadIdx.x;
    for (int i = tid; i < D_DIM * E_NUM; i += blockDim.x) swg[i] = wg[i];
    __syncthreads();

    int t = blockIdx.x * blockDim.x + tid;              // exact grid: B_TOK threads
    const float* xr = x + (size_t)t * D_DIM;
    float lg[E_NUM];
    #pragma unroll
    for (int e = 0; e < E_NUM; ++e) lg[e] = 0.0f;
    for (int d = 0; d < D_DIM; ++d) {
        float xv = xr[d];
        #pragma unroll
        for (int e = 0; e < E_NUM; ++e) lg[e] += xv * swg[d * E_NUM + e];
    }
    // top-1 (first index on ties, matches jax top_k)
    int i0 = 0; float v0 = lg[0];
    #pragma unroll
    for (int e = 1; e < E_NUM; ++e) if (lg[e] > v0) { v0 = lg[e]; i0 = e; }
    // top-2
    int i1 = -1; float v1 = -3.4e38f;
    #pragma unroll
    for (int e = 0; e < E_NUM; ++e) if (e != i0 && lg[e] > v1) { v1 = lg[e]; i1 = e; }
    // softmax over {v0, v1}, T = 1
    float e1 = expf(v1 - v0);
    float g0 = 1.0f / (1.0f + e1);
    float g1 = e1 * g0;

    int p0 = atomicAdd(&cnt[i0], 1);
    idxl[i0 * B_TOK + p0] = t;  gatel[i0 * B_TOK + p0] = g0;
    atomicAdd(&imp[i0], g0);
    int p1 = atomicAdd(&cnt[i1], 1);
    idxl[i1 * B_TOK + p1] = t;  gatel[i1 * B_TOK + p1] = g1;
    atomicAdd(&imp[i1], g1);
}

// ---------------- expert GEMMs via WMMA bf16, M = 32 tokens / block ----------------
__global__ __launch_bounds__(256)
void k_moe(const unsigned* __restrict__ xbf, const unsigned* __restrict__ W1p,
           const unsigned* __restrict__ W2p, const float* __restrict__ b1,
           const float* __restrict__ b2, const int* __restrict__ idxl,
           const float* __restrict__ gatel, const int* __restrict__ cnt,
           float* __restrict__ y) {
    const int e = blockIdx.x / TPE;
    const int t = blockIdx.x % TPE;
    const int c = cnt[e];
    const int base = t * M_TILE;
    if (base >= c) return;                               // uniform early exit

    __shared__ int      s_tok[M_TILE];
    __shared__ float    s_gate[M_TILE];
    __shared__ unsigned s_h32[M_TILE * (H_DIM / 2)];     // 32KB: h tile, bf16 row-major

    const int tid  = threadIdx.x;
    const int lane = tid & 31;
    const int wave = tid >> 5;
    const int ln   = lane & 15;
    const int hi   = (lane >> 4) & 1;
    const int aoff = hi * 16;                            // A-fragment K lane-offset

    if (tid < M_TILE) {
        int j = base + tid;
        int tok = (j < c) ? idxl[e * B_TOK + j] : -1;
        s_tok[tid]  = tok;
        s_gate[tid] = (tok >= 0) ? gatel[e * B_TOK + j] : 0.0f;
    }
    __syncthreads();

    // ---- GEMM1: h(32x512) = A(32x128) x W1[e](128x512), K = 4 steps of 32 ----
    // A fragments for both M-tiles (pad rows read row 0, gated to 0 later)
    v16bf afr[2][4];
    #pragma unroll
    for (int mt = 0; mt < 2; ++mt) {
        int arow = s_tok[mt * 16 + ln]; arow = arow < 0 ? 0 : arow;
        const unsigned* xrow = xbf + (size_t)arow * (D_DIM / 2);
        #pragma unroll
        for (int ks = 0; ks < 4; ++ks) {
            FragU f;
            int o = (32 * ks + aoff) >> 1;
            #pragma unroll
            for (int i = 0; i < 4; ++i) { f.u[i] = xrow[o + i]; f.u[4 + i] = xrow[o + 4 + i]; }
            afr[mt][ks] = f.v;
        }
    }
    const unsigned* w1e = W1p + (size_t)e * ((D_DIM / 2) * H_DIM);
    const v8f vzero = {0.f, 0.f, 0.f, 0.f, 0.f, 0.f, 0.f, 0.f};
    v8f acc[2][4];
    #pragma unroll
    for (int mt = 0; mt < 2; ++mt)
        #pragma unroll
        for (int nt = 0; nt < 4; ++nt) acc[mt][nt] = vzero;

    #pragma unroll
    for (int nt = 0; nt < 4; ++nt) {
        const int ntg = wave * 4 + nt;                   // global N-tile 0..31
        #pragma unroll
        for (int ks = 0; ks < 4; ++ks) {
            FragU f;                                     // one B-frag feeds 2 WMMAs
            const unsigned* p = w1e + (((ntg * 4 + ks) << 8) + lane * 8);
            #pragma unroll
            for (int i = 0; i < 8; ++i) f.u[i] = p[i];   // 2x b128, coalesced
            acc[0][nt] = __builtin_amdgcn_wmma_f32_16x16x32_bf16(
                false, afr[0][ks], false, f.v, (short)0, acc[0][nt], false, false);
            acc[1][nt] = __builtin_amdgcn_wmma_f32_16x16x32_bf16(
                false, afr[1][ks], false, f.v, (short)0, acc[1][nt], false, false);
        }
    }

    // bias + relu + bf16 -> LDS
    unsigned short* s_h16 = (unsigned short*)s_h32;
    #pragma unroll
    for (int nt = 0; nt < 4; ++nt) {
        const int col  = (wave * 4 + nt) * 16 + ln;
        const float bs = b1[e * H_DIM + col];
        #pragma unroll
        for (int mt = 0; mt < 2; ++mt) {
            #pragma unroll
            for (int i = 0; i < 8; ++i) {
                int M = mt * 16 + i + hi * 8;
                float v = acc[mt][nt][i] + bs;
                v = v > 0.f ? v : 0.f;
                s_h16[M * H_DIM + col] = f2bf(v);
            }
        }
    }
    __syncthreads();

    // ---- GEMM2: y(32x128) = h(32x512) x W2[e](512x128), K = 16 steps of 32 ----
    const unsigned* w2e = W2p + (size_t)e * ((H_DIM / 2) * D_DIM);
    const int col2 = wave * 16 + ln;
    v8f acc2[2]; acc2[0] = vzero; acc2[1] = vzero;
    #pragma unroll
    for (int ks = 0; ks < 16; ++ks) {
        FragU fb;                                        // one B-frag feeds 2 WMMAs
        const unsigned* p = w2e + (((wave * 16 + ks) << 8) + lane * 8);
        #pragma unroll
        for (int i = 0; i < 8; ++i) fb.u[i] = p[i];
        #pragma unroll
        for (int mt = 0; mt < 2; ++mt) {
            FragU fa;
            int o = (mt * 16 + ln) * (H_DIM / 2) + ((32 * ks + aoff) >> 1);
            #pragma unroll
            for (int i = 0; i < 4; ++i) { fa.u[i] = s_h32[o + i]; fa.u[4 + i] = s_h32[o + 4 + i]; }
            acc2[mt] = __builtin_amdgcn_wmma_f32_16x16x32_bf16(
                false, fa.v, false, fb.v, (short)0, acc2[mt], false, false);
        }
    }

    const float bs2 = b2[e * D_DIM + col2];
    #pragma unroll
    for (int mt = 0; mt < 2; ++mt) {
        #pragma unroll
        for (int i = 0; i < 8; ++i) {
            int M = mt * 16 + i + hi * 8;
            int tok = s_tok[M];
            if (tok >= 0)
                atomicAdd(&y[(size_t)tok * D_DIM + col2], s_gate[M] * (acc2[mt][i] + bs2));
        }
    }
}

// ---------------- epilogue: y==0 -> EPS ----------------
__global__ void k_eps(float* y) {
    int i = blockIdx.x * blockDim.x + threadIdx.x;
    if (i < BD) {
        float v = y[i];
        y[i] = (v == 0.0f) ? EPS_COMBINE : v;
    }
}

// ---------------- aux loss ----------------
__global__ void k_loss(const float* __restrict__ imp, const int* __restrict__ cnt,
                       float* __restrict__ out) {
    float mi = 0.f, ml = 0.f;
    for (int e = 0; e < E_NUM; ++e) { mi += imp[e]; ml += (float)cnt[e]; }
    mi /= E_NUM; ml /= E_NUM;
    float vi = 0.f, vl = 0.f;
    for (int e = 0; e < E_NUM; ++e) {
        float di = imp[e] - mi;        vi += di * di;
        float dl = (float)cnt[e] - ml; vl += dl * dl;
    }
    vi /= (E_NUM - 1); vl /= (E_NUM - 1);
    out[BD] = vi / (mi * mi + 1e-10f) + vl / (ml * ml + 1e-10f);
}

extern "C" void kernel_launch(void* const* d_in, const int* in_sizes, int n_in,
                              void* d_out, int out_size, void* d_ws, size_t ws_size,
                              hipStream_t stream) {
    (void)in_sizes; (void)n_in; (void)out_size; (void)ws_size;
    const float* x  = (const float*)d_in[0];
    const float* wg = (const float*)d_in[1];
    const float* W1 = (const float*)d_in[2];
    const float* b1 = (const float*)d_in[3];
    const float* W2 = (const float*)d_in[4];
    const float* b2 = (const float*)d_in[5];
    float* y = (float*)d_out;

    // workspace layout
    unsigned* xbf   = (unsigned*)d_ws;                         // BD/2 dwords        (4 MB)
    unsigned* W1p   = xbf + (BD / 2);                          // 262144 dwords      (1 MB)
    unsigned* W2p   = W1p + (E_NUM * (D_DIM / 2) * H_DIM);     // 262144 dwords      (1 MB)
    int*      idxl  = (int*)(W2p + E_NUM * (H_DIM / 2) * D_DIM);   // 8*16384 ints   (512 KB)
    float*    gatel = (float*)(idxl + E_NUM * B_TOK);              // 8*16384 floats (512 KB)
    int*      cnt   = (int*)(gatel + E_NUM * B_TOK);               // 8 ints
    float*    imp   = (float*)(cnt + 16);                          // 8 floats

    k_zero   <<<(BD + 255) / 256, 256, 0, stream>>>(y, cnt, imp);
    k_pack_x <<<(BD / 2) / 256,   256, 0, stream>>>(x, xbf);
    k_pack_w1<<<262144 / 256,     256, 0, stream>>>(W1, W1p);
    k_pack_w2<<<262144 / 256,     256, 0, stream>>>(W2, W2p);
    k_gate   <<<B_TOK / 256,      256, 0, stream>>>(x, wg, cnt, imp, idxl, gatel);
    k_moe    <<<E_NUM * TPE,      256, 0, stream>>>(xbf, W1p, W2p, b1, b2, idxl, gatel, cnt, y);
    k_eps    <<<(BD + 255) / 256, 256, 0, stream>>>(y);
    k_loss   <<<1, 1, 0, stream>>>(imp, cnt, y);
}
